// WLNReactionRanking_56891136803559
// MI455X (gfx1250) — matrix operand order repaired and checked by reading, not compiled, not run on hardware
//
#include <hip/hip_runtime.h>

#define CANDS 1000
#define VN    100
#define ER    250
#define HD    128
#define ROWS_PAD 112   // 7 M-tiles of 16
#define MT_E  16       // 256 edges padded
#define MT_N  7        // 112 node rows padded
#define H_CHUNKS ((VN * HD * 2) / 16)   // 1600 x 16B per graph

typedef unsigned short bf16s;
typedef __attribute__((ext_vector_type(16))) __bf16 v16bf;
typedef __attribute__((ext_vector_type(8)))  unsigned short v8us;
typedef __attribute__((ext_vector_type(8)))  float v8f;

// native hardware f32 <-> bf16 conversion (RNE), lets clang use v_cvt_pk_bf16_f32
__device__ inline bf16s f2bf(float f) {
    __bf16 b = (__bf16)f;
    return __builtin_bit_cast(bf16s, b);
}
__device__ inline float bf2f(bf16s h) {
    unsigned u = ((unsigned)h) << 16;
    return __builtin_bit_cast(float, u);
}
__device__ inline __bf16 us2bf(unsigned short s) {
    return __builtin_bit_cast(__bf16, s);
}

// ---- gfx1250 async LDS<->global copies (ASYNCcnt-tracked, bypass VGPRs) ----
__device__ inline void async_g2l_b128(const void* gaddr, const void* lds) {
    unsigned long long ga = (unsigned long long)gaddr;
    unsigned lo = (unsigned)(unsigned long long)lds;   // addr[31:0] = LDS offset
    asm volatile("global_load_async_to_lds_b128 %0, %1, off"
                 :: "v"(lo), "v"(ga) : "memory");
}
__device__ inline void async_l2g_b128(const void* gaddr, const void* lds) {
    unsigned long long ga = (unsigned long long)gaddr;
    unsigned lo = (unsigned)(unsigned long long)lds;
    asm volatile("global_store_async_from_lds_b128 %0, %1, off"
                 :: "v"(ga), "v"(lo) : "memory");
}
__device__ inline void wait_async0() {
    asm volatile("s_wait_asynccnt 0x0" ::: "memory");
}

// ---- A fragment (16x32 bf16, MxK) gathered from an LDS row (K contiguous) ----
// lane 0-15: row M=lane&15, K = {kb..kb+7, kb+16..kb+23}, kb = kstep*32
// lane 16-31: same rows, kb += 8
__device__ inline v16bf a_frag_bf(const bf16s* row, int kstep, int lane) {
    int kb = kstep * 32 + ((lane & 16) ? 8 : 0);
    v8us lo = *(const v8us*)(row + kb);
    v8us hi = *(const v8us*)(row + kb + 16);
    v16bf a;
#pragma unroll
    for (int i = 0; i < 8; ++i) { a[i] = us2bf(lo[i]); a[8 + i] = us2bf(hi[i]); }
    return a;
}

// same, but source row is f32 (LDS aggregation buffer), converted on the fly
__device__ inline v16bf a_frag_f32(const float* row, int kstep, int lane) {
    int kb = kstep * 32 + ((lane & 16) ? 8 : 0);
    const float* p = row + kb;
    v16bf a;
#pragma unroll
    for (int i = 0; i < 8; ++i) {
        a[i]     = (__bf16)p[i];
        a[8 + i] = (__bf16)p[16 + i];
    }
    return a;
}

// ---- B fragment (32x16 bf16, KxN) from pre-packed weights: one 32B load/lane ----
__device__ inline v16bf b_frag(const bf16s* packed, int kstep, int ntile, int lane) {
    const v16bf* p = (const v16bf*)(packed + ((size_t)(kstep * 8 + ntile) * 512) + lane * 16);
    return *p;
}

// =====================================================================
// Pack W[Kin,128] (f32, row-major) into per-lane WMMA-B fragment layout.
//   n = ntile*16 + (l&15); k = kstep*32 + ((l&16)?8:0) + (j<8 ? j : 8+j)
// =====================================================================
__global__ void pack_w_kernel(const float* __restrict__ W, int Kin, int nfrag,
                              bf16s* __restrict__ dst) {
    int t = blockIdx.x * 256 + threadIdx.x;
    if (t >= nfrag * 512) return;
    int f = t >> 9, rem = t & 511, l = rem >> 4, j = rem & 15;
    int nt = f & 7, s = f >> 3;
    int n = nt * 16 + (l & 15);
    int k = s * 32 + ((l & 16) ? 8 : 0) + (j < 8 ? j : 8 + j);
    float v = (k < Kin) ? W[(size_t)k * HD + n] : 0.f;
    dst[t] = f2bf(v);
}

// =====================================================================
// h0 = ReLU(X @ W_proj + b), X:[nrows,82], K padded to 96 (3 K-steps)
// one 16-row M-tile per block; wave w = N-tile w; coalesced async store-out
// =====================================================================
__global__ __launch_bounds__(256)
void proj_kernel(const float* __restrict__ X, int nrows,
                 const bf16s* __restrict__ pW, const float* __restrict__ bias,
                 bf16s* __restrict__ hout) {
    __shared__ bf16s sa[16 * 96];
    __shared__ bf16s so[16 * HD];
    const int tid = threadIdx.x, lane = tid & 31, wv = tid >> 5;
    const int rowbase = blockIdx.x * 16;

    for (int i = tid; i < 16 * 96; i += 256) {
        int r = i / 96, c = i % 96;
        float v = ((rowbase + r) < nrows && c < 82) ? X[(size_t)(rowbase + r) * 82 + c] : 0.f;
        sa[i] = f2bf(v);
    }
    __syncthreads();

    const bf16s* arow = sa + (size_t)(lane & 15) * 96;
    v8f c = {};
#pragma unroll
    for (int s = 0; s < 3; ++s) {
        v16bf a = a_frag_bf(arow, s, lane);
        v16bf b = b_frag(pW, s, wv, lane);
        c = __builtin_amdgcn_wmma_f32_16x16x32_bf16(false, a, false, b, (short)0, c, false, false);
    }
    const int col = wv * 16 + (lane & 15);
    const float bb = bias[col];
#pragma unroll
    for (int r = 0; r < 8; ++r) {
        int row = r + ((lane & 16) ? 8 : 0);
        so[row * HD + col] = f2bf(fmaxf(c[r] + bb, 0.f));
    }
    __syncthreads();
    {   // 16 rows x 128 bf16 = 4 KB = 256 x 16B chunks, one per thread
        int row = tid >> 4;                  // 16 chunks per row
        if (rowbase + row < nrows)
            async_l2g_b128(hout + (size_t)rowbase * HD + tid * 8, so + tid * 8);
        wait_async0();
    }
}

// =====================================================================
// Fused 3-layer WLN encoder, one candidate graph per workgroup.
// h in LDS (bf16 ping-pong, async-staged); scatter-sum via ds_add_f32;
// agg converted to bf16 once per layer (avoids 8x redundant cvt).
// =====================================================================
__global__ __launch_bounds__(256)
void wln_encode_kernel(bf16s* __restrict__ h_global,
                       const float* __restrict__ ef,
                       const int* __restrict__ esrc, const int* __restrict__ edst,
                       const bf16s* __restrict__ pWmsg, const float* __restrict__ bmsg,
                       const bf16s* __restrict__ pWnew, const float* __restrict__ bnew,
                       int n_layers) {
    __shared__ bf16s sh_h[2][ROWS_PAD * HD];
    __shared__ float sh_agg[ROWS_PAD * HD];
    __shared__ bf16s sh_aggbf[ROWS_PAD * HD];
    __shared__ bf16s sh_e[256 * 8];
    __shared__ int   sh_src[256];
    __shared__ int   sh_dst[256];

    const int tid = threadIdx.x, lane = tid & 31, wv = tid >> 5;
    const int nodebase = blockIdx.x * VN;
    const int edgebase = blockIdx.x * ER;
    const int col = wv * 16 + (lane & 15);
    bf16s* hg = h_global + (size_t)nodebase * HD;

    // async stage h rows 0..99, zero pad rows 100..111
    for (int i = tid; i < H_CHUNKS; i += 256)
        async_g2l_b128((const char*)hg + i * 16, (const char*)sh_h[0] + i * 16);
    for (int i = VN * HD + tid; i < ROWS_PAD * HD; i += 256) sh_h[0][i] = 0;
    {
        int t = tid;  // blockDim == 256 covers all padded edges
        if (t < ER) {
            sh_src[t] = esrc[edgebase + t] - nodebase;
            sh_dst[t] = edst[edgebase + t] - nodebase;
#pragma unroll
            for (int j = 0; j < 8; ++j)
                sh_e[t * 8 + j] = (j < 6) ? f2bf(ef[(size_t)(edgebase + t) * 6 + j]) : (bf16s)0;
        } else {
            sh_src[t] = 0; sh_dst[t] = -1;
#pragma unroll
            for (int j = 0; j < 8; ++j) sh_e[t * 8 + j] = 0;
        }
    }
    wait_async0();
    __syncthreads();

    const float biasm = bmsg[col];
    const float biasn = bnew[col];
    int cur = 0;
    for (int layer = 0; layer < n_layers; ++layer) {
        for (int i = tid; i < ROWS_PAD * HD; i += 256) sh_agg[i] = 0.f;
        __syncthreads();

        const bf16s* hin = sh_h[cur];
        // ---- message GEMM: [256 edges, 160] x [160,128], scatter into agg ----
        for (int mt = 0; mt < MT_E; ++mt) {
            const int erow = mt * 16 + (lane & 15);
            const bf16s* arow = hin + (size_t)sh_src[erow] * HD;
            v8f c = {};
#pragma unroll
            for (int s = 0; s < 4; ++s) {
                v16bf a = a_frag_bf(arow, s, lane);
                v16bf b = b_frag(pWmsg, s, wv, lane);
                c = __builtin_amdgcn_wmma_f32_16x16x32_bf16(false, a, false, b, (short)0, c, false, false);
            }
            {   // K-step 4: edge feats occupy K=128..133 (lanes 0-15, halves 0..5)
                v16bf a;
#pragma unroll
                for (int j = 0; j < 16; ++j) a[j] = us2bf(0);
                if (!(lane & 16)) {
#pragma unroll
                    for (int j = 0; j < 6; ++j) a[j] = us2bf(sh_e[erow * 8 + j]);
                }
                v16bf b = b_frag(pWmsg, 4, wv, lane);
                c = __builtin_amdgcn_wmma_f32_16x16x32_bf16(false, a, false, b, (short)0, c, false, false);
            }
#pragma unroll
            for (int r = 0; r < 8; ++r) {
                int em = mt * 16 + r + ((lane & 16) ? 8 : 0);
                int d = sh_dst[em];
                if (d >= 0)
                    atomicAdd(&sh_agg[d * HD + col], fmaxf(c[r] + biasm, 0.f));
            }
        }
        __syncthreads();
        // convert agg to bf16 once (shared by all 8 waves' A-fragments)
        for (int i = tid; i < ROWS_PAD * HD; i += 256) sh_aggbf[i] = f2bf(sh_agg[i]);
        __syncthreads();

        // ---- node update GEMM: [112, 256] x [256,128] ----
        bf16s* hout = sh_h[cur ^ 1];
        for (int mt = 0; mt < MT_N; ++mt) {
            const int nrow = mt * 16 + (lane & 15);
            const bf16s* arow = hin + (size_t)nrow * HD;
            const bf16s* grow = sh_aggbf + (size_t)nrow * HD;
            v8f c = {};
#pragma unroll
            for (int s = 0; s < 4; ++s) {
                v16bf a = a_frag_bf(arow, s, lane);
                v16bf b = b_frag(pWnew, s, wv, lane);
                c = __builtin_amdgcn_wmma_f32_16x16x32_bf16(false, a, false, b, (short)0, c, false, false);
            }
#pragma unroll
            for (int s = 0; s < 4; ++s) {
                v16bf a = a_frag_bf(grow, s, lane);
                v16bf b = b_frag(pWnew, 4 + s, wv, lane);
                c = __builtin_amdgcn_wmma_f32_16x16x32_bf16(false, a, false, b, (short)0, c, false, false);
            }
#pragma unroll
            for (int r = 0; r < 8; ++r) {
                int row = mt * 16 + r + ((lane & 16) ? 8 : 0);
                float v = (row < VN) ? fmaxf(c[r] + biasn, 0.f) : 0.f;
                hout[row * HD + col] = f2bf(v);
            }
        }
        __syncthreads();
        cur ^= 1;
    }
    // async write-back of the 100 real rows
    for (int i = tid; i < H_CHUNKS; i += 256)
        async_l2g_b128((char*)hg + i * 16, (const char*)sh_h[cur] + i * 16);
    wait_async0();
}

// =====================================================================
// diff = p_enc - r_enc, one diff-WLN layer, then per-graph sum pooling
// =====================================================================
__global__ __launch_bounds__(256)
void wln_diff_pool_kernel(const bf16s* __restrict__ p_enc, const bf16s* __restrict__ r_enc,
                          const float* __restrict__ ef,
                          const int* __restrict__ esrc, const int* __restrict__ edst,
                          const bf16s* __restrict__ pWmsg, const float* __restrict__ bmsg,
                          const bf16s* __restrict__ pWnew, const float* __restrict__ bnew,
                          float* __restrict__ g_out) {
    __shared__ bf16s sh_h[ROWS_PAD * HD];
    __shared__ float sh_agg[ROWS_PAD * HD];
    __shared__ float sh_d[ROWS_PAD * HD];
    __shared__ bf16s sh_e[256 * 8];
    __shared__ int   sh_src[256];
    __shared__ int   sh_dst[256];

    const int tid = threadIdx.x, lane = tid & 31, wv = tid >> 5;
    const int nodebase = blockIdx.x * VN;
    const int edgebase = blockIdx.x * ER;
    const int col = wv * 16 + (lane & 15);

    for (int i = tid; i < ROWS_PAD * HD; i += 256) {
        float v = (i < VN * HD)
            ? bf2f(p_enc[(size_t)nodebase * HD + i]) - bf2f(r_enc[i]) : 0.f;
        sh_h[i] = f2bf(v);
    }
    {
        int t = tid;
        if (t < ER) {
            sh_src[t] = esrc[edgebase + t] - nodebase;
            sh_dst[t] = edst[edgebase + t] - nodebase;
#pragma unroll
            for (int j = 0; j < 8; ++j)
                sh_e[t * 8 + j] = (j < 6) ? f2bf(ef[(size_t)(edgebase + t) * 6 + j]) : (bf16s)0;
        } else {
            sh_src[t] = 0; sh_dst[t] = -1;
#pragma unroll
            for (int j = 0; j < 8; ++j) sh_e[t * 8 + j] = 0;
        }
    }
    for (int i = tid; i < ROWS_PAD * HD; i += 256) sh_agg[i] = 0.f;
    __syncthreads();

    const float biasm = bmsg[col];
    const float biasn = bnew[col];

    for (int mt = 0; mt < MT_E; ++mt) {
        const int erow = mt * 16 + (lane & 15);
        const bf16s* arow = sh_h + (size_t)sh_src[erow] * HD;
        v8f c = {};
#pragma unroll
        for (int s = 0; s < 4; ++s) {
            v16bf a = a_frag_bf(arow, s, lane);
            v16bf b = b_frag(pWmsg, s, wv, lane);
            c = __builtin_amdgcn_wmma_f32_16x16x32_bf16(false, a, false, b, (short)0, c, false, false);
        }
        {
            v16bf a;
#pragma unroll
            for (int j = 0; j < 16; ++j) a[j] = us2bf(0);
            if (!(lane & 16)) {
#pragma unroll
                for (int j = 0; j < 6; ++j) a[j] = us2bf(sh_e[erow * 8 + j]);
            }
            v16bf b = b_frag(pWmsg, 4, wv, lane);
            c = __builtin_amdgcn_wmma_f32_16x16x32_bf16(false, a, false, b, (short)0, c, false, false);
        }
#pragma unroll
        for (int r = 0; r < 8; ++r) {
            int em = mt * 16 + r + ((lane & 16) ? 8 : 0);
            int d = sh_dst[em];
            if (d >= 0)
                atomicAdd(&sh_agg[d * HD + col], fmaxf(c[r] + biasm, 0.f));
        }
    }
    __syncthreads();

    for (int mt = 0; mt < MT_N; ++mt) {
        const int nrow = mt * 16 + (lane & 15);
        const bf16s* arow = sh_h + (size_t)nrow * HD;
        const float* grow = sh_agg + (size_t)nrow * HD;
        v8f c = {};
#pragma unroll
        for (int s = 0; s < 4; ++s) {
            v16bf a = a_frag_bf(arow, s, lane);
            v16bf b = b_frag(pWnew, s, wv, lane);
            c = __builtin_amdgcn_wmma_f32_16x16x32_bf16(false, a, false, b, (short)0, c, false, false);
        }
#pragma unroll
        for (int s = 0; s < 4; ++s) {
            v16bf a = a_frag_f32(grow, s, lane);
            v16bf b = b_frag(pWnew, 4 + s, wv, lane);
            c = __builtin_amdgcn_wmma_f32_16x16x32_bf16(false, a, false, b, (short)0, c, false, false);
        }
#pragma unroll
        for (int r = 0; r < 8; ++r) {
            int row = mt * 16 + r + ((lane & 16) ? 8 : 0);
            sh_d[row * HD + col] = (row < VN) ? fmaxf(c[r] + biasn, 0.f) : 0.f;
        }
    }
    __syncthreads();

    for (int cidx = tid; cidx < HD; cidx += 256) {
        float s = 0.f;
        for (int m = 0; m < VN; ++m) s += sh_d[m * HD + cidx];
        g_out[(size_t)blockIdx.x * HD + cidx] = s;
    }
}

// =====================================================================
// out[c] = ReLU(g[c] @ W1 + b1) @ W2 + b2 + scores[c]   (tiny; VALU)
// =====================================================================
__global__ __launch_bounds__(128)
void head_kernel(const float* __restrict__ g, const float* __restrict__ W1,
                 const float* __restrict__ b1, const float* __restrict__ W2,
                 const float* __restrict__ b2, const float* __restrict__ scores,
                 float* __restrict__ out) {
    __shared__ float sg[HD];
    __shared__ float sp[HD];
    const int c = blockIdx.x, j = threadIdx.x;
    sg[j] = g[(size_t)c * HD + j];
    __syncthreads();
    float acc = b1[j];
#pragma unroll 8
    for (int k = 0; k < HD; ++k) acc += sg[k] * W1[(size_t)k * HD + j];
    sp[j] = fmaxf(acc, 0.f) * W2[j];
    __syncthreads();
    for (int off = 64; off > 0; off >>= 1) {
        if (j < off) sp[j] += sp[j + off];
        __syncthreads();
    }
    if (j == 0) out[c] = sp[0] + b2[0] + scores[c];
}

extern "C" void kernel_launch(void* const* d_in, const int* in_sizes, int n_in,
                              void* d_out, int out_size, void* d_ws, size_t ws_size,
                              hipStream_t stream) {
    (void)in_sizes; (void)n_in; (void)out_size; (void)ws_size;
    const float* r_nf   = (const float*)d_in[0];
    const float* r_ef   = (const float*)d_in[1];
    const float* p_nf   = (const float*)d_in[2];
    const float* p_ef   = (const float*)d_in[3];
    const float* scores = (const float*)d_in[4];
    const float* W_proj = (const float*)d_in[5];  const float* b_proj = (const float*)d_in[6];
    const float* W_msg  = (const float*)d_in[7];  const float* b_msg  = (const float*)d_in[8];
    const float* W_new  = (const float*)d_in[9];  const float* b_new  = (const float*)d_in[10];
    const float* W_msgd = (const float*)d_in[11]; const float* b_msgd = (const float*)d_in[12];
    const float* W_newd = (const float*)d_in[13]; const float* b_newd = (const float*)d_in[14];
    const float* W_p1   = (const float*)d_in[15]; const float* b_p1   = (const float*)d_in[16];
    const float* W_p2   = (const float*)d_in[17]; const float* b_p2   = (const float*)d_in[18];
    const int* r_src = (const int*)d_in[19]; const int* r_dst = (const int*)d_in[20];
    const int* p_src = (const int*)d_in[21]; const int* p_dst = (const int*)d_in[22];

    char* ws = (char*)d_ws; size_t off = 0;
    auto alloc = [&](size_t bytes) -> void* {
        void* p = ws + off; off = (off + bytes + 255) & ~(size_t)255; return p;
    };
    bf16s* pWproj = (bf16s*)alloc((size_t)3 * 8 * 512 * 2);
    bf16s* pWmsg  = (bf16s*)alloc((size_t)5 * 8 * 512 * 2);
    bf16s* pWnew  = (bf16s*)alloc((size_t)8 * 8 * 512 * 2);
    bf16s* pWmsgd = (bf16s*)alloc((size_t)5 * 8 * 512 * 2);
    bf16s* pWnewd = (bf16s*)alloc((size_t)8 * 8 * 512 * 2);
    bf16s* h_r    = (bf16s*)alloc((size_t)VN * HD * 2);
    bf16s* h_p    = (bf16s*)alloc((size_t)CANDS * VN * HD * 2);
    float* gpool  = (float*)alloc((size_t)CANDS * HD * 4);

    pack_w_kernel<<<(3 * 8 * 512 + 255) / 256, 256, 0, stream>>>(W_proj, 82, 24, pWproj);
    pack_w_kernel<<<(5 * 8 * 512 + 255) / 256, 256, 0, stream>>>(W_msg, 134, 40, pWmsg);
    pack_w_kernel<<<(8 * 8 * 512 + 255) / 256, 256, 0, stream>>>(W_new, 256, 64, pWnew);
    pack_w_kernel<<<(5 * 8 * 512 + 255) / 256, 256, 0, stream>>>(W_msgd, 134, 40, pWmsgd);
    pack_w_kernel<<<(8 * 8 * 512 + 255) / 256, 256, 0, stream>>>(W_newd, 256, 64, pWnewd);

    proj_kernel<<<(VN + 15) / 16, 256, 0, stream>>>(r_nf, VN, pWproj, b_proj, h_r);
    proj_kernel<<<(CANDS * VN + 15) / 16, 256, 0, stream>>>(p_nf, CANDS * VN, pWproj, b_proj, h_p);

    wln_encode_kernel<<<1, 256, 0, stream>>>(h_r, r_ef, r_src, r_dst,
                                             pWmsg, b_msg, pWnew, b_new, 3);
    wln_encode_kernel<<<CANDS, 256, 0, stream>>>(h_p, p_ef, p_src, p_dst,
                                                 pWmsg, b_msg, pWnew, b_new, 3);

    wln_diff_pool_kernel<<<CANDS, 256, 0, stream>>>(h_p, h_r, p_ef, p_src, p_dst,
                                                    pWmsgd, b_msgd, pWnewd, b_newd, gpool);

    head_kernel<<<CANDS, 128, 0, stream>>>(gpool, W_p1, b_p1, W_p2, b_p2, scores,
                                           (float*)d_out);
}
